// ConversationGNN_26903675142307
// MI455X (gfx1250) — matrix operand order repaired and checked by reading, not compiled, not run on hardware
//
#include <hip/hip_runtime.h>
#include <hip/hip_bf16.h>
#include <stdint.h>

// ---------------------------------------------------------------------------
// Types for CDNA5 WMMA (gfx1250, wave32)
// ---------------------------------------------------------------------------
typedef __attribute__((ext_vector_type(16))) __bf16 v16bf;
typedef __attribute__((ext_vector_type(8)))  __bf16 bf16x8;
typedef __attribute__((ext_vector_type(2)))  __bf16 bf16x2;
typedef __attribute__((ext_vector_type(8)))  float  v8f;
typedef __attribute__((ext_vector_type(4)))  float  f32x4;
typedef __attribute__((ext_vector_type(4)))  unsigned u32x4;

#define NEG_SLOPE 0.2f

// Pack two f32 -> one u32 holding (lo, hi) bf16.
// The clang builtin __builtin_amdgcn_cvt_pk_bf16_f32 is not declared on this
// toolchain, but the hardware has V_CVT_PK_BF16_F32 — use inline asm.
__device__ __forceinline__ unsigned packbf(float lo, float hi) {
  unsigned r;
  asm volatile("v_cvt_pk_bf16_f32 %0, %1, %2" : "=v"(r) : "v"(lo), "v"(hi));
  return r;
}

// ---------------------------------------------------------------------------
// Generic fill
// ---------------------------------------------------------------------------
__global__ void fill_f32(float* __restrict__ p, float v, long long n) {
  long long i = (long long)blockIdx.x * blockDim.x + threadIdx.x;
  if (i < n) p[i] = v;
}

// ---------------------------------------------------------------------------
// Tiled bf16 WMMA GEMM:  C[M,Nc] = A[M,K] * B[K,Nc] (+ A2[M,2]*B2[2,Nc]) (+bias)
// Block = 256 threads = 8 waves; 64(M) x 128(N) tile; K stepped by 32.
// Each wave: one A fragment reused over FOUR v_wmma_f32_16x16x32_bf16.
// Double-buffered LDS (two barriers / K-step), register-staged global loads.
// Requires: Nc % 128 == 0, K % 32 == 0  (384/1024/128 all OK).
// ---------------------------------------------------------------------------
__global__ __launch_bounds__(256)
void gemm_bf16_wmma(const float* __restrict__ A, const float* __restrict__ B,
                    const float* __restrict__ bias,
                    const float* __restrict__ A2, const float* __restrict__ B2,
                    float* __restrict__ C, int M, int K, int Nc)
{
  // bf16 tiles packed as u32 pairs:  A: 64x32 (4KB)   B^T: 128x32 (8KB)
  __shared__ __align__(64) unsigned As32[2][64 * 16];
  __shared__ __align__(64) unsigned Bs32[2][128 * 16];

  const int tid  = threadIdx.x;
  const int lane = tid & 31;
  const int wave = tid >> 5;       // 0..7
  const int wm   = wave & 3;       // 16-row slab
  const int wn   = wave >> 2;      // 64-col slab (0..1)

  const int block_m = blockIdx.y * 64;
  const int block_n = blockIdx.x * 128;

  // staging thread mapping
  const int a_m  = tid >> 2;            // 0..63
  const int a_k  = (tid & 3) * 8;       // 0,8,16,24
  const int a_gm = block_m + a_m;
  const int b_k  = (tid >> 4) * 2;      // 0,2,..,30
  const int b_n  = (tid & 15) * 8;      // 0..120

  v8f acc[4] = {v8f{}, v8f{}, v8f{}, v8f{}};

  f32x4 ar0, ar1;                 // A staging regs (8 floats)
  f32x4 br00, br01, br10, br11;   // B staging regs (2 k-rows x 8 n)

  // ---- preload tile k0 = 0
  {
    if (a_gm < M) {
      const float* s = A + (size_t)a_gm * K + a_k;
      ar0 = *(const f32x4*)(s);
      ar1 = *(const f32x4*)(s + 4);
    } else {
      ar0 = f32x4{0.f, 0.f, 0.f, 0.f};
      ar1 = ar0;
    }
    const float* s0 = B + (size_t)b_k * Nc + block_n + b_n;
    br00 = *(const f32x4*)(s0);
    br01 = *(const f32x4*)(s0 + 4);
    br10 = *(const f32x4*)(s0 + Nc);
    br11 = *(const f32x4*)(s0 + Nc + 4);
  }

  int cur = 0;
  for (int k0 = 0; k0 < K; k0 += 32) {
    // ---- commit staged registers to LDS[cur]
    {
      u32x4 w;
      w.x = packbf(ar0.x, ar0.y);
      w.y = packbf(ar0.z, ar0.w);
      w.z = packbf(ar1.x, ar1.y);
      w.w = packbf(ar1.z, ar1.w);
      *(u32x4*)&As32[cur][a_m * 16 + (a_k >> 1)] = w;   // ds_store_b128
    }
    {
      unsigned* bp = &Bs32[cur][(b_k >> 1)];
#pragma unroll
      for (int i = 0; i < 4; ++i) {
        bp[(b_n + i)     * 16] = packbf(br00[i], br10[i]);
        bp[(b_n + i + 4) * 16] = packbf(br01[i], br11[i]);
      }
    }

    // ---- issue global loads for next tile (in flight during WMMAs)
    const int kn = k0 + 32;
    if (kn < K) {
      if (a_gm < M) {
        const float* s = A + (size_t)a_gm * K + kn + a_k;
        ar0 = *(const f32x4*)(s);
        ar1 = *(const f32x4*)(s + 4);
      } else {
        ar0 = f32x4{0.f, 0.f, 0.f, 0.f};
        ar1 = ar0;
      }
      const float* s0 = B + (size_t)(kn + b_k) * Nc + block_n + b_n;
      br00 = *(const f32x4*)(s0);
      br01 = *(const f32x4*)(s0 + 4);
      br10 = *(const f32x4*)(s0 + Nc);
      br11 = *(const f32x4*)(s0 + Nc + 4);
    }

    __syncthreads();

    // ---- fragments from LDS[cur]
    const __bf16* As = (const __bf16*)&As32[cur][0];
    const __bf16* Bs = (const __bf16*)&Bs32[cur][0];

    // A fragment (ISA 16-bit A 16x32 layout): lanes<16 K=0..7,16..23 ; hi K=8..15,24..31
    const int mrow = wm * 16 + (lane & 15);
    const int kb   = (lane >> 4) * 8;
    bf16x8 a0 = *(const bf16x8*)&As[mrow * 32 + kb];
    bf16x8 a1 = *(const bf16x8*)&As[mrow * 32 + 16 + kb];
    v16bf afrag = __builtin_shufflevector(a0, a1, 0, 1, 2, 3, 4, 5, 6, 7,
                                          8, 9, 10, 11, 12, 13, 14, 15);

    // B fragments (32x16): N = lane&15 ; same K split as A: lanes<16 K=0..7,16..23 ; hi K=8..15,24..31
    const int kbB = (lane >> 4) * 8;
    const int ncb = wn * 64 + (lane & 15);
#pragma unroll
    for (int t = 0; t < 4; ++t) {
      bf16x8 b0 = *(const bf16x8*)&Bs[(ncb + t * 16) * 32 + kbB];
      bf16x8 b1 = *(const bf16x8*)&Bs[(ncb + t * 16) * 32 + 16 + kbB];
      v16bf bfrag = __builtin_shufflevector(b0, b1, 0, 1, 2, 3, 4, 5, 6, 7,
                                            8, 9, 10, 11, 12, 13, 14, 15);
      acc[t] = __builtin_amdgcn_wmma_f32_16x16x32_bf16(false, afrag, false, bfrag,
                                                       (short)0, acc[t], false, false);
    }

    __syncthreads();
    cur ^= 1;
  }

  // ---- epilogue (f32 C/D layout: VGPR i -> row i (+8 hi lanes), N = lane&15)
  const int row0 = block_m + wm * 16 + ((lane >> 4) ? 8 : 0);
  const int colb = block_n + wn * 64 + (lane & 15);
#pragma unroll
  for (int i = 0; i < 8; ++i) {
    int gm = row0 + i;
    if (gm < M) {
      float na0 = 0.f, na1 = 0.f;
      if (A2) {
        na0 = A2[(size_t)gm * 2 + 0];
        na1 = A2[(size_t)gm * 2 + 1];
      }
#pragma unroll
      for (int t = 0; t < 4; ++t) {
        int col = colb + t * 16;
        float c = acc[t][i];
        if (A2)   c += na0 * B2[col] + na1 * B2[Nc + col];
        if (bias) c += bias[col];
        C[(size_t)gm * Nc + col] = c;
      }
    }
  }
}

// ---------------------------------------------------------------------------
// Per-(node,head) attention projections: alpha_s/d[n,h] = dot(h[n,h,:], a_s/d[h,:])
// ---------------------------------------------------------------------------
__global__ void alpha_kernel(const float* __restrict__ h,
                             const float* __restrict__ a_src,
                             const float* __restrict__ a_dst,
                             float* __restrict__ out_s, float* __restrict__ out_d,
                             int Nn, int H, int C)
{
  int node = blockIdx.x;
  if (node >= Nn) return;
  int hh   = threadIdx.x >> 5;
  int lane = threadIdx.x & 31;
  const float* hp = h + (size_t)node * H * C + hh * C;
  float s = 0.f, d = 0.f;
  for (int c = lane; c < C; c += 32) {
    float v = hp[c];
    s += v * a_src[hh * C + c];
    d += v * a_dst[hh * C + c];
  }
#pragma unroll
  for (int off = 16; off > 0; off >>= 1) {
    s += __shfl_down(s, off, 32);
    d += __shfl_down(d, off, 32);
  }
  if (lane == 0) {
    out_s[node * H + hh] = s;
    out_d[node * H + hh] = d;
  }
}

// ---------------------------------------------------------------------------
// Edge pipeline. Edge id e in [0, E+Nn): e < E -> graph edge, else self loop.
// H is a power of two; hshift = log2(H).
// ---------------------------------------------------------------------------
__device__ __forceinline__ void edge_sd(const long long* __restrict__ ei,
                                        int E, long long e,
                                        long long& s, long long& d) {
  if (e < E) { s = ei[e]; d = ei[E + e]; }
  else       { s = d = e - E; }
}

__global__ void edge_score_max(const long long* __restrict__ ei, int E, int Nn,
                               const float* __restrict__ as, const float* __restrict__ ad,
                               float* __restrict__ ebuf, float* __restrict__ mx, int hshift)
{
  long long idx = (long long)blockIdx.x * blockDim.x + threadIdx.x;
  long long tot = (long long)(E + Nn) << hshift;
  if (idx >= tot) return;
  long long e = idx >> hshift;
  int hh = (int)(idx & ((1 << hshift) - 1));
  long long s, d; edge_sd(ei, E, e, s, d);
  float v = as[(s << hshift) + hh] + ad[(d << hshift) + hh];
  v = (v >= 0.f) ? v : NEG_SLOPE * v;           // LeakyReLU(0.2)
  ebuf[idx] = v;
  atomicMax(&mx[(d << hshift) + hh], v);        // GLOBAL_ATOMIC_MAX_NUM_F32
}

__global__ void edge_exp_sum(const long long* __restrict__ ei, int E, int Nn,
                             float* __restrict__ ebuf, const float* __restrict__ mx,
                             float* __restrict__ dn, int hshift)
{
  long long idx = (long long)blockIdx.x * blockDim.x + threadIdx.x;
  long long tot = (long long)(E + Nn) << hshift;
  if (idx >= tot) return;
  long long e = idx >> hshift;
  int hh = (int)(idx & ((1 << hshift) - 1));
  long long s, d; edge_sd(ei, E, e, s, d);
  (void)s;
  float v = __expf(ebuf[idx] - mx[(d << hshift) + hh]);
  ebuf[idx] = v;
  atomicAdd(&dn[(d << hshift) + hh], v);
}

// One block per edge: out[dst,h,c] += h[src,h,c] * alpha(e,h). cshift = log2(C).
// Vectorized: each lane handles 4 consecutive channels (b128 load + 4 atomics).
__global__ void edge_aggregate(const long long* __restrict__ ei, int E, int Nn,
                               const float* __restrict__ h,
                               const float* __restrict__ ebuf,
                               const float* __restrict__ dn,
                               float* __restrict__ out, int hshift, int cshift)
{
  long long e = blockIdx.x;
  long long s, d; edge_sd(ei, E, e, s, d);
  const int HC = 1 << (hshift + cshift);
  const float* hs = h   + (size_t)s * HC;
  float*       od = out + (size_t)d * HC;
  for (int i4 = threadIdx.x * 4; i4 < HC; i4 += blockDim.x * 4) {
    int hh = i4 >> cshift;   // 4 consecutive channels share a head (C >= 128)
    float alpha = ebuf[(e << hshift) + hh] / (dn[(d << hshift) + hh] + 1e-16f);
    f32x4 hv = *(const f32x4*)&hs[i4];
    atomicAdd(&od[i4 + 0], hv.x * alpha);       // GLOBAL_ATOMIC_ADD_F32
    atomicAdd(&od[i4 + 1], hv.y * alpha);
    atomicAdd(&od[i4 + 2], hv.z * alpha);
    atomicAdd(&od[i4 + 3], hv.w * alpha);
  }
}

// out[n, c] += bias[c]; optional ELU. colmask = Cw-1 (Cw power of two).
__global__ void bias_act(float* __restrict__ p, const float* __restrict__ bias,
                         long long total, int colmask, int do_elu)
{
  long long i = (long long)blockIdx.x * blockDim.x + threadIdx.x;
  if (i >= total) return;
  float v = p[i] + bias[(int)(i & colmask)];
  if (do_elu) v = (v > 0.f) ? v : (__expf(v) - 1.0f);
  p[i] = v;
}

// ---------------------------------------------------------------------------
// Host-side launcher
// ---------------------------------------------------------------------------
extern "C" void kernel_launch(void* const* d_in, const int* in_sizes, int n_in,
                              void* d_out, int out_size, void* d_ws, size_t ws_size,
                              hipStream_t stream)
{
  const float*     x         = (const float*)d_in[0];
  const float*     node_attr = (const float*)d_in[1];
  const long long* ei        = (const long long*)d_in[2];   // int64 [2,E]
  const float*     enc_W     = (const float*)d_in[3];       // [386,384]
  const float*     enc_b     = (const float*)d_in[4];
  const float*     W1        = (const float*)d_in[5];       // [384,1024]
  const float*     a_src1    = (const float*)d_in[6];       // [4,256]
  const float*     a_dst1    = (const float*)d_in[7];
  const float*     b1        = (const float*)d_in[8];       // [1024]
  const float*     W2        = (const float*)d_in[9];       // [1024,128]
  const float*     a_src2    = (const float*)d_in[10];      // [1,128]
  const float*     a_dst2    = (const float*)d_in[11];
  const float*     b2        = (const float*)d_in[12];      // [128]

  const int N  = in_sizes[0] / 384;
  const int E  = in_sizes[2] / 2;
  const int EE = E + N;   // edges + self loops

  // ---- workspace carve
  char* ws = (char*)d_ws;
  size_t off = 0;
  auto carve = [&](size_t bytes) -> float* {
    float* p = (float*)(ws + off);
    off += (bytes + 255) & ~(size_t)255;
    return p;
  };
  float* h0  = carve((size_t)N * 384  * 4);
  float* h1  = carve((size_t)N * 1024 * 4);
  float* o1  = carve((size_t)N * 1024 * 4);
  float* h2  = carve((size_t)N * 128  * 4);
  float* as1 = carve((size_t)N * 4 * 4);
  float* ad1 = carve((size_t)N * 4 * 4);
  float* m1  = carve((size_t)N * 4 * 4);
  float* dn1 = carve((size_t)N * 4 * 4);
  float* as2 = carve((size_t)N * 4);
  float* ad2 = carve((size_t)N * 4);
  float* m2  = carve((size_t)N * 4);
  float* dn2 = carve((size_t)N * 4);
  float* eb1 = carve((size_t)EE * 4 * 4);
  float* eb2 = carve((size_t)EE * 4);
  (void)ws_size; (void)n_in; (void)out_size;

  auto cdiv = [](long long a, long long b) -> int { return (int)((a + b - 1) / b); };

  // ---- init accumulators
  fill_f32<<<cdiv((long long)N * 4, 256), 256, 0, stream>>>(m1, -3.0e38f, (long long)N * 4);
  fill_f32<<<cdiv((long long)N * 4, 256), 256, 0, stream>>>(dn1, 0.0f, (long long)N * 4);
  fill_f32<<<cdiv((long long)N * 1024, 256), 256, 0, stream>>>(o1, 0.0f, (long long)N * 1024);
  fill_f32<<<cdiv((long long)N, 256), 256, 0, stream>>>(m2, -3.0e38f, (long long)N);
  fill_f32<<<cdiv((long long)N, 256), 256, 0, stream>>>(dn2, 0.0f, (long long)N);
  fill_f32<<<cdiv((long long)N * 128, 256), 256, 0, stream>>>((float*)d_out, 0.0f, (long long)N * 128);

  // ---- node encoder: h0 = [x | node_attr] @ enc_W + enc_b
  gemm_bf16_wmma<<<dim3(384 / 128, cdiv(N, 64)), 256, 0, stream>>>(
      x, enc_W, enc_b, node_attr, enc_W + (size_t)384 * 384, h0, N, 384, 384);

  // ---- GAT layer 1 ------------------------------------------------------
  gemm_bf16_wmma<<<dim3(1024 / 128, cdiv(N, 64)), 256, 0, stream>>>(
      h0, W1, nullptr, nullptr, nullptr, h1, N, 384, 1024);
  alpha_kernel<<<N, 128, 0, stream>>>(h1, a_src1, a_dst1, as1, ad1, N, 4, 256);
  edge_score_max<<<cdiv((long long)EE * 4, 256), 256, 0, stream>>>(ei, E, N, as1, ad1, eb1, m1, 2);
  edge_exp_sum  <<<cdiv((long long)EE * 4, 256), 256, 0, stream>>>(ei, E, N, eb1, m1, dn1, 2);
  edge_aggregate<<<EE, 256, 0, stream>>>(ei, E, N, h1, eb1, dn1, o1, 2, 8);   // H=4,C=256
  bias_act<<<cdiv((long long)N * 1024, 256), 256, 0, stream>>>(o1, b1, (long long)N * 1024, 1023, 1);

  // ---- GAT layer 2 ------------------------------------------------------
  gemm_bf16_wmma<<<dim3(128 / 128, cdiv(N, 64)), 256, 0, stream>>>(
      o1, W2, nullptr, nullptr, nullptr, h2, N, 1024, 128);
  alpha_kernel<<<N, 32, 0, stream>>>(h2, a_src2, a_dst2, as2, ad2, N, 1, 128);
  edge_score_max<<<cdiv((long long)EE, 256), 256, 0, stream>>>(ei, E, N, as2, ad2, eb2, m2, 0);
  edge_exp_sum  <<<cdiv((long long)EE, 256), 256, 0, stream>>>(ei, E, N, eb2, m2, dn2, 0);
  edge_aggregate<<<EE, 32, 0, stream>>>(ei, E, N, h2, eb2, dn2, (float*)d_out, 0, 7); // H=1,C=128
  bias_act<<<cdiv((long long)N * 128, 256), 256, 0, stream>>>((float*)d_out, b2, (long long)N * 128, 127, 0);
}
